// LearnableSpectralDistanceLayer_28939489641146
// MI455X (gfx1250) — matrix operand order repaired and checked by reading, not compiled, not run on hardware
//
#include <hip/hip_runtime.h>
#include <hip/hip_bf16.h>
#include <math.h>

#define B_ 32
#define N_ 512
#define H_ 128
#define BT 128      // block tile (M and N)
#define KT 32       // K step = one bf16 WMMA K
#define LDB 40      // padded LDS row stride (bf16 elems): 80B rows, banks spread

constexpr float INF_DIST = 512.0f;
constexpr float EPS_ = 1e-6f;

typedef __attribute__((ext_vector_type(16))) __bf16 v16bf;
typedef __attribute__((ext_vector_type(8)))  __bf16 v8bf;
typedef __attribute__((ext_vector_type(4)))  __bf16 v4bf;
typedef __attribute__((ext_vector_type(8)))  float  v8f;

// ---------------- phi(lambda): 2-layer MLP with exact GELU ----------------
__global__ void phi_kernel(const float* __restrict__ eigvals,
                           const float* __restrict__ w1,
                           const float* __restrict__ b1,
                           const float* __restrict__ w2,
                           const float* __restrict__ b2,
                           float* __restrict__ phi) {
  int idx = blockIdx.x * 256 + threadIdx.x;        // 0 .. B*N-1
  float ev = eigvals[idx];
  float acc = 0.0f;
  #pragma unroll 4
  for (int h = 0; h < H_; ++h) {
    float x = fmaf(ev, w1[h], b1[h]);
    float g = 0.5f * x * (1.0f + erff(x * 0.70710678118654752f));  // exact GELU
    acc = fmaf(g, w2[h], acc);
  }
  phi[idx] = acc + b2[0];
}

// ---------------- diag(L) in full fp32: d[b,i] = sum_k V[i,k]^2 * phi[k] ----
__global__ void diag_kernel(const float* __restrict__ V,
                            const float* __restrict__ phi,
                            float* __restrict__ diag) {
  int wave = threadIdx.x >> 5;
  int lane = threadIdx.x & 31;
  int row  = blockIdx.x * 8 + wave;                // 0 .. B*N-1
  const float* vr = V + (size_t)row * N_;
  const float* ph = phi + (size_t)(row >> 9) * N_; // row / N_ == batch
  float acc = 0.0f;
  #pragma unroll
  for (int k = lane; k < N_; k += 32) {
    float v = vr[k];
    acc = fmaf(v * v, ph[k], acc);
  }
  #pragma unroll
  for (int m = 16; m >= 1; m >>= 1) acc += __shfl_xor(acc, m, 32);
  if (lane == 0) diag[row] = acc;
}

// Split a scaled float4 into bf16 hi/lo quads (converted ONCE per block here).
__device__ __forceinline__ void split4(const float4 f, v4bf* hi, v4bf* lo) {
  float vv[4] = {f.x, f.y, f.z, f.w};
  #pragma unroll
  for (int e = 0; e < 4; ++e) {
    __bf16 hb = (__bf16)vv[e];
    (*hi)[e] = hb;
    (*lo)[e] = (__bf16)(vv[e] - (float)hb);
  }
}

// Concat two 8-elem bf16 LDS loads into one 16-elem fragment (register-only).
__device__ __forceinline__ v16bf cat16(v8bf a, v8bf b) {
  return __builtin_shufflevector(a, b, 0,1,2,3,4,5,6,7,8,9,10,11,12,13,14,15);
}

// ------- fused GEMM (bf16x3 split-precision WMMA) + distance epilogue -------
__global__ __launch_bounds__(256)
void gemm_dist_kernel(const float* __restrict__ V,
                      const float* __restrict__ phi,
                      const float* __restrict__ diag,
                      const float* __restrict__ mask,
                      float* __restrict__ out) {
  // bf16 hi/lo tiles staged in LDS: conversion happens once per element/block.
  __shared__ __align__(16) __bf16 ldsAhi[BT * LDB];
  __shared__ __align__(16) __bf16 ldsAlo[BT * LDB];
  __shared__ __align__(16) __bf16 ldsBhi[BT * LDB];
  __shared__ __align__(16) __bf16 ldsBlo[BT * LDB];

  const int b = blockIdx.z;
  const int rowBase = blockIdx.y * BT;
  const int colBase = blockIdx.x * BT;
  const int t    = threadIdx.x;
  const int wave = t >> 5, lane = t & 31;
  const int waveM = wave >> 2, waveN = wave & 3;   // 2x4 wave grid
  const int mOrig = waveM * 64;                    // wave covers 64 rows
  const int nOrig = waveN * 32;                    // wave covers 32 cols
  const int half  = lane >> 4, lan16 = lane & 15;

  const float* Vb = V   + (size_t)b * N_ * N_;
  const float* ph = phi + (size_t)b * N_;

  v8f zero = {};
  v8f acc[4][2];
  #pragma unroll
  for (int m = 0; m < 4; ++m)
    #pragma unroll
    for (int n = 0; n < 2; ++n) acc[m][n] = zero;

  for (int kt = 0; kt < N_ / KT; ++kt) {
    const int kBase = kt * KT;
    __syncthreads();
    // Stage + split tiles: 128 rows x 32 floats each -> bf16 hi/lo in LDS.
    #pragma unroll
    for (int i = 0; i < 4; ++i) {
      int idx = t + i * 256;
      int r  = idx >> 3;
      int k0 = (idx & 7) * 4;
      float4 a4 = *(const float4*)&Vb[(size_t)(rowBase + r) * N_ + kBase + k0];
      a4.x *= ph[kBase + k0 + 0];
      a4.y *= ph[kBase + k0 + 1];
      a4.z *= ph[kBase + k0 + 2];
      a4.w *= ph[kBase + k0 + 3];
      v4bf hi, lo;
      split4(a4, &hi, &lo);
      *(v4bf*)&ldsAhi[r * LDB + k0] = hi;
      *(v4bf*)&ldsAlo[r * LDB + k0] = lo;
      float4 b4 = *(const float4*)&Vb[(size_t)(colBase + r) * N_ + kBase + k0];
      split4(b4, &hi, &lo);
      *(v4bf*)&ldsBhi[r * LDB + k0] = hi;
      *(v4bf*)&ldsBlo[r * LDB + k0] = lo;
    }
    if (kt + 1 < N_ / KT) {                        // warm L2/WGP$ for next K tile
      int r = t >> 3, k0 = (t & 7) * 4;
      __builtin_prefetch(&Vb[(size_t)(rowBase + r) * N_ + kBase + KT + k0], 0, 0);
      __builtin_prefetch(&Vb[(size_t)(colBase + r) * N_ + kBase + KT + k0], 0, 0);
    }
    __syncthreads();

    // B fragments (32x16, K-major): lane holds col n=lan16, K = half*16 + e.
    // 16 contiguous bf16 at row*LDB + half*16 -> two 16B ds loads, no VALU.
    v16bf bhi[2], blo[2];
    #pragma unroll
    for (int n = 0; n < 2; ++n) {
      int off = (nOrig + n * 16 + lan16) * LDB + half * 16;
      bhi[n] = cat16(*(const v8bf*)&ldsBhi[off], *(const v8bf*)&ldsBhi[off + 8]);
      blo[n] = cat16(*(const v8bf*)&ldsBlo[off], *(const v8bf*)&ldsBlo[off + 8]);
    }

    #pragma unroll
    for (int m = 0; m < 4; ++m) {
      // A fragment (16x32): lane holds row m=lan16, K = half*16 + e.
      int roff = (mOrig + m * 16 + lan16) * LDB + half * 16;
      v16bf ahi = cat16(*(const v8bf*)&ldsAhi[roff],
                        *(const v8bf*)&ldsAhi[roff + 8]);
      v16bf alo = cat16(*(const v8bf*)&ldsAlo[roff],
                        *(const v8bf*)&ldsAlo[roff + 8]);
      #pragma unroll
      for (int n = 0; n < 2; ++n) {   // bf16x3 split-precision accumulate
        acc[m][n] = __builtin_amdgcn_wmma_f32_16x16x32_bf16(
            false, alo, false, bhi[n], (short)0, acc[m][n], false, false);
        acc[m][n] = __builtin_amdgcn_wmma_f32_16x16x32_bf16(
            false, ahi, false, blo[n], (short)0, acc[m][n], false, false);
        acc[m][n] = __builtin_amdgcn_wmma_f32_16x16x32_bf16(
            false, ahi, false, bhi[n], (short)0, acc[m][n], false, false);
      }
    }
  }

  // Fused distance epilogue: L never touches HBM.
  const float* db = diag + (size_t)b * N_;
  const float* mb = mask + (size_t)b * N_ * N_;
  float*       ob = out  + (size_t)b * N_ * N_;
  #pragma unroll
  for (int m = 0; m < 4; ++m) {
    #pragma unroll
    for (int n = 0; n < 2; ++n) {
      int gj = colBase + nOrig + n * 16 + lan16;   // C/D layout: col = lane&15
      float dj = db[gj];
      #pragma unroll
      for (int r = 0; r < 8; ++r) {                // row = (lane>>4)*8 + r
        int gi = rowBase + mOrig + m * 16 + half * 8 + r;
        float Lij = acc[m][n][r];
        float sq  = db[gi] + dj - 2.0f * Lij;
        float dst = sqrtf(fmaxf(sq, EPS_));
        float cm  = mb[(size_t)gi * N_ + gj];
        ob[(size_t)gi * N_ + gj] = cm * dst + INF_DIST * (1.0f - cm);
      }
    }
  }
}

extern "C" void kernel_launch(void* const* d_in, const int* in_sizes, int n_in,
                              void* d_out, int out_size, void* d_ws, size_t ws_size,
                              hipStream_t stream) {
  const float* eigvecs = (const float*)d_in[0];  // [B,N,N]
  const float* eigvals = (const float*)d_in[1];  // [B,N]
  const float* w1      = (const float*)d_in[2];  // [1,H]
  const float* b1      = (const float*)d_in[3];  // [H]
  const float* w2      = (const float*)d_in[4];  // [H,1]
  const float* b2      = (const float*)d_in[5];  // [1]
  const float* cmask   = (const float*)d_in[6];  // [B,N,N]
  float* out = (float*)d_out;

  float* phi  = (float*)d_ws;                    // B*N floats
  float* diag = phi + (size_t)B_ * N_;           // B*N floats

  phi_kernel<<<(B_ * N_) / 256, 256, 0, stream>>>(eigvals, w1, b1, w2, b2, phi);
  diag_kernel<<<(B_ * N_) / 8, 256, 0, stream>>>(eigvecs, phi, diag);
  dim3 grid(N_ / BT, N_ / BT, B_);
  gemm_dist_kernel<<<grid, 256, 0, stream>>>(eigvecs, phi, diag, cmask, out);
}
